// FrequencyPromptModule_33165737460321
// MI455X (gfx1250) — compile-verified, matrix operand order checked by reading, not run on hardware
//
#include <hip/hip_runtime.h>
#include <hip/hip_bf16.h>
#include <cmath>

typedef __attribute__((ext_vector_type(16))) _Float16 v16h;
typedef __attribute__((ext_vector_type(8)))  float    v8f;

#define NIMG 8192
#define NCH  200
#define HW   81      // 9*9
#define EMB  64

// cos/sin(2*pi*t/9), t=0..8
__device__ __constant__ float C9[9] = {
    1.0f,  0.76604444f,  0.17364818f, -0.5f, -0.93969262f,
   -0.93969262f, -0.5f,  0.17364818f,  0.76604444f};
__device__ __constant__ float S9[9] = {
    0.0f,  0.64278761f,  0.98480775f,  0.86602540f,  0.34202014f,
   -0.34202014f, -0.86602540f, -0.98480775f, -0.64278761f};
// after fftshift, kept rows/cols 2..5 correspond to freq indices (s+5)%9:
__device__ __constant__ int KSEL[4] = {7, 8, 0, 1};

// ---------------------------------------------------------------------------
// Kernel 1: channel mean (memory-bound 530 MB stream) + 16 DFT bins -> feats
// One block per image; 8 waves x 25 channels; coalesced b32 loads.
// x is read exactly once and is larger than the 192 MB L2 -> non-temporal.
// ---------------------------------------------------------------------------
__global__ __launch_bounds__(256)
void mean_dft_kernel(const float* __restrict__ x, float* __restrict__ feats) {
    __shared__ float img[HW];
    const int b    = blockIdx.x;
    const int tid  = threadIdx.x;
    const int wave = tid >> 5;
    const int lane = tid & 31;

    if (tid < HW) img[tid] = 0.0f;
    __syncthreads();

    const float* xb = x + (size_t)b * (NCH * HW);
    float s0 = 0.0f, s1 = 0.0f, s2 = 0.0f;
    const int c0 = wave * 25;                 // 200 = 8 waves * 25 channels
    for (int c = c0; c < c0 + 25; ++c) {
        const float* p = xb + c * HW;
        s0 += __builtin_nontemporal_load(p + lane);
        s1 += __builtin_nontemporal_load(p + lane + 32);
        if (lane < HW - 64) s2 += __builtin_nontemporal_load(p + lane + 64);
    }
    atomicAdd(&img[lane],      s0);
    atomicAdd(&img[lane + 32], s1);
    if (lane < HW - 64) atomicAdd(&img[lane + 64], s2);
    __syncthreads();

    if (tid < 16) {
        const int ky = KSEL[tid >> 2];
        const int kx = KSEL[tid & 3];
        float re = 0.0f, im = 0.0f;
        for (int y = 0; y < 9; ++y) {
            const int ry = (ky * y) % 9;
            for (int xx = 0; xx < 9; ++xx) {
                int t = ry + (kx * xx) % 9;
                if (t >= 9) t -= 9;
                const float v = img[y * 9 + xx];
                re += v * C9[t];
                im -= v * S9[t];
            }
        }
        // |FFT(mean)| = |FFT(sum)| / NCH
        feats[b * 16 + tid] = sqrtf(re * re + im * im) * (1.0f / NCH);
    }
}

// ---------------------------------------------------------------------------
// Kernel 2: MLP via V_WMMA_F32_16X16X32_F16. One wave handles 16 rows.
// ---------------------------------------------------------------------------
// K index held by half-element e (0..15) of a 16-bit A/B fragment:
__device__ __forceinline__ int kmap(int e, int hi) {
    const int vg = e >> 1, sub = e & 1;
    return ((vg >> 2) << 4) + ((vg & 3) << 1) + sub + (hi ? 8 : 0);
}

__global__ __launch_bounds__(32)
void mlp_wmma_kernel(const float* __restrict__ feats,
                     const float* __restrict__ W1, const float* __restrict__ b1,
                     const float* __restrict__ W2, const float* __restrict__ b2v,
                     float* __restrict__ out) {
    __shared__ _Float16 hbuf[16 * EMB];       // hidden activations, A-relayout
    const int lane = threadIdx.x;
    const int hi   = lane >> 4;               // 0 or 1
    const int m    = lane & 15;
    const int b0   = blockIdx.x * 16;

    // A1 = feats rows b0..b0+15, K padded 16 -> 32 with zeros
    v16h a1;
    for (int e = 0; e < 16; ++e) {
        const int K = kmap(e, hi);
        a1[e] = (K < 16) ? (_Float16)feats[(b0 + m) * 16 + K] : (_Float16)0.0f;
    }

    // H = gelu(feats @ W1 + b1), staged to LDS as fp16
    for (int nt = 0; nt < 4; ++nt) {
        v16h bf;
        for (int e = 0; e < 16; ++e) {
            const int K = kmap(e, hi);
            bf[e] = (K < 16) ? (_Float16)W1[K * EMB + nt * 16 + m] : (_Float16)0.0f;
        }
        v8f c = {};
        c = __builtin_amdgcn_wmma_f32_16x16x32_f16(false, a1, false, bf,
                                                   (short)0, c, false, false);
        const int col = nt * 16 + m;
        const float bias = b1[col];
        for (int r = 0; r < 8; ++r) {
            const int row = r + hi * 8;
            const float v = c[r] + bias;
            const float g = 0.5f * v * (1.0f + erff(v * 0.70710678118f));
            hbuf[row * EMB + col] = (_Float16)g;
        }
    }
    __syncthreads();

    // A2 = H (16 x 64) in two K=32 chunks
    v16h a2lo, a2hi2;
    for (int e = 0; e < 16; ++e) {
        const int K = kmap(e, hi);
        a2lo[e]  = hbuf[m * EMB + K];
        a2hi2[e] = hbuf[m * EMB + 32 + K];
    }

    // out = H @ W2 + b2
    for (int nt = 0; nt < 4; ++nt) {
        v8f acc = {};
        v16h bf0, bf1;
        for (int e = 0; e < 16; ++e) {
            const int K = kmap(e, hi);
            bf0[e] = (_Float16)W2[K * EMB + nt * 16 + m];
            bf1[e] = (_Float16)W2[(K + 32) * EMB + nt * 16 + m];
        }
        acc = __builtin_amdgcn_wmma_f32_16x16x32_f16(false, a2lo,  false, bf0,
                                                     (short)0, acc, false, false);
        acc = __builtin_amdgcn_wmma_f32_16x16x32_f16(false, a2hi2, false, bf1,
                                                     (short)0, acc, false, false);
        const int col = nt * 16 + m;
        const float bias = b2v[col];
        for (int r = 0; r < 8; ++r) {
            const int row = r + hi * 8;
            out[(size_t)(b0 + row) * EMB + col] = acc[r] + bias;
        }
    }
}

extern "C" void kernel_launch(void* const* d_in, const int* in_sizes, int n_in,
                              void* d_out, int out_size, void* d_ws, size_t ws_size,
                              hipStream_t stream) {
    const float* x  = (const float*)d_in[0];
    const float* W1 = (const float*)d_in[1];
    const float* b1 = (const float*)d_in[2];
    const float* W2 = (const float*)d_in[3];
    const float* b2 = (const float*)d_in[4];
    float* out   = (float*)d_out;
    float* feats = (float*)d_ws;              // 8192*16 floats = 512 KB scratch

    mean_dft_kernel<<<NIMG, 256, 0, stream>>>(x, feats);
    mlp_wmma_kernel<<<NIMG / 16, 32, 0, stream>>>(feats, W1, b1, W2, b2, out);
}
